// GRUModel_11304353923873
// MI455X (gfx1250) — matrix operand order repaired and checked by reading, not compiled, not run on hardware
//
#include <hip/hip_runtime.h>
#include <hip/hip_bf16.h>

typedef __attribute__((ext_vector_type(16))) _Float16 v16h;
typedef __attribute__((ext_vector_type(8)))  float    v8f;

// ---------------- problem constants ----------------
constexpr int Bsz = 4096, T = 256, F = 32, H = 50, MLPD = 25, Cout = 4;
constexpr int NTG = 12;                // 3 gates * 4 N-tiles of 16 (gate padded 50->64)
constexpr int WAVES = 4;               // waves per block (wave32)
constexpr int ROWS_PER_BLOCK = WAVES * 16;
constexpr int NBLOCKS = Bsz / ROWS_PER_BLOCK;   // 64

// fragment-ready weight sizes (halfs): NT * KC * 32 lanes * 16 halfs
constexpr int HS_WIH0 = NTG * 1 * 512;   // K=32  (F)
constexpr int HS_WHH0 = NTG * 2 * 512;   // K=64  (Hpad)
constexpr int HS_WIH1 = NTG * 2 * 512;
constexpr int HS_WHH1 = NTG * 2 * 512;
constexpr int HS_W1   = 2 * 2 * 512;     // MLP: 25->32 out, K 50->64
constexpr int HS_W2   = 1 * 1 * 512;     // 4->16 out, K 25->32
constexpr int HS_TOTAL = HS_WIH0 + HS_WHH0 + HS_WIH1 + HS_WHH1 + HS_W1 + HS_W2;
constexpr int FL_BIAS  = 4 * 192 + 32 + 16;      // bx0,bh0,bx1,bh1,b1,b2
constexpr int FL_H     = WAVES * 2 * 1024;       // h1[16][64] + h2[16][64] per wave
constexpr int SMEM_BYTES = HS_TOTAL * 2 + (FL_BIAS + FL_H) * 4;   // 127,168 B

// ---------------- device helpers ----------------
__device__ __forceinline__ v8f wmma_acc(v16h a, v16h b, v8f c) {
  return __builtin_amdgcn_wmma_f32_16x16x32_f16(false, a, false, b, (short)0, c,
                                                false, false);
}
__device__ __forceinline__ v8f wmma_zero(v16h a, v16h b) {
  v8f z = {};   // SRC2 inline 0 -> no C-init movs
  return __builtin_amdgcn_wmma_f32_16x16x32_f16(false, a, false, b, (short)0, z,
                                                false, false);
}

// branch-free activations on the TRANS pipe (v_exp_f32 / v_rcp_f32 / v_tanh_f32)
__device__ __forceinline__ float fast_sigmoid(float x) {
  const float t = __builtin_amdgcn_exp2f(x * -1.442695040889f);
  return __builtin_amdgcn_rcpf(1.0f + t);
}
__device__ __forceinline__ float fast_tanh(float x) {
#if __has_builtin(__builtin_amdgcn_tanhf)
  return __builtin_amdgcn_tanhf(x);
#else
  const float xc = fminf(fmaxf(x, -30.0f), 30.0f);
  const float t = __builtin_amdgcn_exp2f(xc * 2.885390081778f);   // 2^(2x*log2e)
  return 1.0f - 2.0f * __builtin_amdgcn_rcpf(t + 1.0f);
#endif
}

// B fragment: lane holds 16 contiguous halfs in frag-ready layout
__device__ __forceinline__ v16h bload(const _Float16* frag, int n, int kc, int KC,
                                      int lane) {
  return *(const v16h*)(frag + (((n * KC) + kc) * 32 + lane) * 16);
}

// A fragment (16x32 f16) from an LDS f32 buffer with row stride `stride`
__device__ __forceinline__ v16h afrag(const float* buf, int stride, int kc, int lane) {
  const int row = lane & 15;
  const int koff = (lane & 16) ? 8 : 0;
  const float* p = buf + row * stride + kc * 32 + koff;
  v16h a;
#pragma unroll
  for (int i = 0; i < 8; ++i) {
    a[i]     = (_Float16)p[i];        // K = koff + i
    a[i + 8] = (_Float16)p[i + 16];   // K = koff + 16 + i
  }
  return a;
}

// raw x_t tile (load early, convert at use -> hides global latency)
struct XRaw { float4 a, b, c, d; };
__device__ __forceinline__ XRaw xload(const float* xp, int lane) {
  const int koff = (lane & 16) ? 8 : 0;
  XRaw r;
  r.a = *(const float4*)(xp + koff);
  r.b = *(const float4*)(xp + koff + 4);
  r.c = *(const float4*)(xp + koff + 16);
  r.d = *(const float4*)(xp + koff + 20);
  return r;
}
__device__ __forceinline__ v16h xcvt(const XRaw& r) {
  v16h a;
  a[0]  = (_Float16)r.a.x; a[1]  = (_Float16)r.a.y; a[2]  = (_Float16)r.a.z; a[3]  = (_Float16)r.a.w;
  a[4]  = (_Float16)r.b.x; a[5]  = (_Float16)r.b.y; a[6]  = (_Float16)r.b.z; a[7]  = (_Float16)r.b.w;
  a[8]  = (_Float16)r.c.x; a[9]  = (_Float16)r.c.y; a[10] = (_Float16)r.c.z; a[11] = (_Float16)r.c.w;
  a[12] = (_Float16)r.d.x; a[13] = (_Float16)r.d.y; a[14] = (_Float16)r.d.z; a[15] = (_Float16)r.d.w;
  return a;
}

// Build frag-ready f16 GRU weight from global f32 W[3H][Din] (W^T indexed [K][N])
__device__ void fill_gru_frag(_Float16* frag, const float* W, int Din, int KC,
                              int tid, int nth) {
  const int total = NTG * KC * 512;
  for (int idx = tid; idx < total; idx += nth) {
    const int i    = idx & 15;
    const int lane = (idx >> 4) & 31;
    const int rem  = idx >> 9;           // n*KC + kc
    const int kc   = rem % KC;
    const int n    = rem / KC;
    const int N = n * 16 + (lane & 15);  // gate-padded column 0..191
    const int K = kc * 32 + ((lane & 16) >> 1) + (i & 7) + ((i & 8) << 1);
    const int g = N >> 6, j = N & 63;
    float v = 0.0f;
    if (j < H && K < Din) v = W[(g * H + j) * Din + K];
    frag[idx] = (_Float16)v;
  }
}

__device__ void fill_mlp_frag(_Float16* frag, const float* W, int Out, int Din,
                              int NT, int KC, int tid, int nth) {
  const int total = NT * KC * 512;
  for (int idx = tid; idx < total; idx += nth) {
    const int i    = idx & 15;
    const int lane = (idx >> 4) & 31;
    const int rem  = idx >> 9;
    const int kc   = rem % KC;
    const int n    = rem / KC;
    const int N = n * 16 + (lane & 15);
    const int K = kc * 32 + ((lane & 16) >> 1) + (i & 7) + ((i & 8) << 1);
    float v = 0.0f;
    if (N < Out && K < Din) v = W[N * Din + K];
    frag[idx] = (_Float16)v;
  }
}

// per-lane, time-invariant gate biases held in VGPRs
struct GateBias {
  float brz[4], bzz[4], bxn[4], bhn[4];
};

__device__ __forceinline__ GateBias make_bias(const float* bx, const float* bh,
                                              int nc) {
  GateBias g;
#pragma unroll
  for (int k = 0; k < 4; ++k) {
    const int col = k * 16 + nc;
    g.brz[k] = bx[col] + bh[col];
    g.bzz[k] = bx[64 + col] + bh[64 + col];
    g.bxn[k] = bx[128 + col];
    g.bhn[k] = bh[128 + col];
  }
  return g;
}

// One GRU step for a 16-row tile. KCX = # K-chunks of the input A fragments.
// hA0/hA1: this layer's hidden-state A fragments (built by caller).
template <int KCX>
__device__ __forceinline__ void gru_step(const v16h* xA, v16h hA0, v16h hA1,
                                         const _Float16* ih, const _Float16* hh,
                                         const GateBias& gb, float* hbuf, int lane,
                                         int hi8) {
  const int nc = lane & 15;
#pragma unroll
  for (int k = 0; k < 4; ++k) {
    const int col = k * 16 + nc;
    // ---- preload ALL B fragments for this tile (one ds clause, waits stagger) ----
    const v16h Bxr0 = bload(ih, k, 0, KCX, lane);
    const v16h Bxz0 = bload(ih, 4 + k, 0, KCX, lane);
    const v16h Bxn0 = bload(ih, 8 + k, 0, KCX, lane);
    const v16h Bhr0 = bload(hh, k, 0, 2, lane);
    const v16h Bhz0 = bload(hh, 4 + k, 0, 2, lane);
    const v16h Bhn0 = bload(hh, 8 + k, 0, 2, lane);
    const v16h Bhr1 = bload(hh, k, 1, 2, lane);
    const v16h Bhz1 = bload(hh, 4 + k, 1, 2, lane);
    const v16h Bhn1 = bload(hh, 8 + k, 1, 2, lane);
    v8f cr, cz, cxn, chn;
    if constexpr (KCX == 2) {
      const v16h Bxr1 = bload(ih, k, 1, 2, lane);
      const v16h Bxz1 = bload(ih, 4 + k, 1, 2, lane);
      const v16h Bxn1 = bload(ih, 8 + k, 1, 2, lane);
      // interleaved accumulator chains: no WMMA->WMMA RAW back-to-back
      cr  = wmma_zero(xA[0], Bxr0);
      cz  = wmma_zero(xA[0], Bxz0);
      cxn = wmma_zero(xA[0], Bxn0);
      chn = wmma_zero(hA0, Bhn0);
      cr  = wmma_acc(hA0, Bhr0, cr);
      cz  = wmma_acc(hA0, Bhz0, cz);
      cxn = wmma_acc(xA[1], Bxn1, cxn);
      chn = wmma_acc(hA1, Bhn1, chn);
      cr  = wmma_acc(xA[1], Bxr1, cr);
      cz  = wmma_acc(xA[1], Bxz1, cz);
      cr  = wmma_acc(hA1, Bhr1, cr);
      cz  = wmma_acc(hA1, Bhz1, cz);
    } else {
      cr  = wmma_zero(xA[0], Bxr0);
      cz  = wmma_zero(xA[0], Bxz0);
      cxn = wmma_zero(xA[0], Bxn0);
      chn = wmma_zero(hA0, Bhn0);
      cr  = wmma_acc(hA0, Bhr0, cr);
      cz  = wmma_acc(hA0, Bhz0, cz);
      chn = wmma_acc(hA1, Bhn1, chn);
      cr  = wmma_acc(hA1, Bhr1, cr);
      cz  = wmma_acc(hA1, Bhz1, cz);
    }
    float* hcol = hbuf + col;
#pragma unroll
    for (int i = 0; i < 8; ++i) {
      const float r = fast_sigmoid(cr[i] + gb.brz[k]);
      const float z = fast_sigmoid(cz[i] + gb.bzz[k]);
      const float nn = fast_tanh((cxn[i] + gb.bxn[k]) + r * (chn[i] + gb.bhn[k]));
      const float hold = hcol[(i + hi8) * 64];
      const float hnew = nn + z * (hold - nn);   // (1-z)*n + z*h
      hcol[(i + hi8) * 64] = hnew;               // padded cols stay exactly 0
    }
  }
}

// ---------------- fused kernel ----------------
__global__ __launch_bounds__(WAVES * 32, 1) void gru2_mlp_fused(
    const float* __restrict__ x, const float* __restrict__ Wih0,
    const float* __restrict__ Whh0, const float* __restrict__ bih0,
    const float* __restrict__ bhh0, const float* __restrict__ Wih1,
    const float* __restrict__ Whh1, const float* __restrict__ bih1,
    const float* __restrict__ bhh1, const float* __restrict__ W1,
    const float* __restrict__ b1, const float* __restrict__ W2,
    const float* __restrict__ b2, float* __restrict__ out) {
  extern __shared__ __align__(32) char smem[];
  _Float16* wih0 = (_Float16*)smem;
  _Float16* whh0 = wih0 + HS_WIH0;
  _Float16* wih1 = whh0 + HS_WHH0;
  _Float16* whh1 = wih1 + HS_WIH1;
  _Float16* w1f  = whh1 + HS_WHH1;
  _Float16* w2f  = w1f + HS_W1;
  float* bx0 = (float*)(w2f + HS_W2);
  float* bh0 = bx0 + 192;
  float* bx1 = bh0 + 192;
  float* bh1 = bx1 + 192;
  float* b1p = bh1 + 192;
  float* b2p = b1p + 32;
  float* hbase = b2p + 16;

  const int tid = threadIdx.x, nth = blockDim.x;

  // ---- one-time weight staging into frag-ready f16 LDS ----
  fill_gru_frag(wih0, Wih0, F, 1, tid, nth);
  fill_gru_frag(whh0, Whh0, H, 2, tid, nth);
  fill_gru_frag(wih1, Wih1, H, 2, tid, nth);
  fill_gru_frag(whh1, Whh1, H, 2, tid, nth);
  fill_mlp_frag(w1f, W1, MLPD, H, 2, 2, tid, nth);
  fill_mlp_frag(w2f, W2, Cout, MLPD, 1, 1, tid, nth);
  for (int idx = tid; idx < 192; idx += nth) {
    const int g = idx >> 6, j = idx & 63;
    float vx0 = 0, vh0 = 0, vx1 = 0, vh1 = 0;
    if (j < H) {
      const int r = g * H + j;
      vx0 = bih0[r]; vh0 = bhh0[r]; vx1 = bih1[r]; vh1 = bhh1[r];
    }
    bx0[idx] = vx0; bh0[idx] = vh0; bx1[idx] = vx1; bh1[idx] = vh1;
  }
  for (int idx = tid; idx < 32; idx += nth) b1p[idx] = (idx < MLPD) ? b1[idx] : 0.0f;
  for (int idx = tid; idx < 16; idx += nth) b2p[idx] = (idx < Cout) ? b2[idx] : 0.0f;
  for (int idx = tid; idx < FL_H; idx += nth) hbase[idx] = 0.0f;  // h1=h2=0
  __syncthreads();

  const int lane = tid & 31;
  const int wave = tid >> 5;
  const int nc = lane & 15;
  const int hi8 = (lane & 16) ? 8 : 0;
  float* h1buf = hbase + wave * 2048;   // [16][64] f32
  float* h2buf = h1buf + 1024;          // [16][64] f32
  const int rowbase = blockIdx.x * ROWS_PER_BLOCK + wave * 16;
  const float* xrow = x + (size_t)(rowbase + nc) * T * F;

  // time-invariant per-lane gate biases -> registers
  const GateBias gb0 = make_bias(bx0, bh0, nc);
  const GateBias gb1 = make_bias(bx1, bh1, nc);

  // hidden-state A fragments carried across iterations (h==0 initially)
  v16h h1A0 = {}, h1A1 = {}, h2A0 = {}, h2A1 = {};

  XRaw xr = xload(xrow, lane);                     // t = 0, issued early
  for (int t = 0; t < T; ++t) {
    const int tn = (t + 1 < T) ? (t + 1) : t;
    XRaw xrn = xload(xrow + tn * F, lane);         // t+1, converted next iter
    __builtin_prefetch(xrow + ((t + 4 < T) ? (t + 4) : t) * F, 0, 3);
    v16h xA[2];
    xA[0] = xcvt(xr);
    gru_step<1>(xA, h1A0, h1A1, wih0, whh0, gb0, h1buf, lane, hi8);
    // layer-1 input frags == next step's layer-0 state frags (both are h1(t))
    v16h hin[2];
    hin[0] = afrag(h1buf, 64, 0, lane);
    hin[1] = afrag(h1buf, 64, 1, lane);
    h1A0 = hin[0]; h1A1 = hin[1];
    gru_step<2>(hin, h2A0, h2A1, wih1, whh1, gb1, h2buf, lane, hi8);
    // rebuild h2 frags now; latency overlaps next iteration's layer-0 work
    h2A0 = afrag(h2buf, 64, 0, lane);
    h2A1 = afrag(h2buf, 64, 1, lane);
    xr = xrn;
  }

  // ---- MLP head on h2[:, T-1, :] (A fragments already in registers) ----
  float* hid = h1buf;  // reuse as [16][32]
#pragma unroll
  for (int n = 0; n < 2; ++n) {
    v8f c = wmma_zero(h2A0, bload(w1f, n, 0, 2, lane));
    c = wmma_acc(h2A1, bload(w1f, n, 1, 2, lane), c);
    const float bb = b1p[n * 16 + nc];
#pragma unroll
    for (int i = 0; i < 8; ++i)
      hid[(i + hi8) * 32 + n * 16 + nc] = fmaxf(c[i] + bb, 0.0f);  // relu
  }
  const v16h ah = afrag(hid, 32, 0, lane);
  v8f c = wmma_zero(ah, bload(w2f, 0, 0, 1, lane));
  const float b2v = b2p[nc];
  if (nc < Cout) {
#pragma unroll
    for (int i = 0; i < 8; ++i)
      out[(size_t)(rowbase + i + hi8) * Cout + nc] = c[i] + b2v;
  }
}

// ---------------- host launch ----------------
extern "C" void kernel_launch(void* const* d_in, const int* in_sizes, int n_in,
                              void* d_out, int out_size, void* d_ws, size_t ws_size,
                              hipStream_t stream) {
  (void)in_sizes; (void)n_in; (void)out_size; (void)d_ws; (void)ws_size;
  const float* x    = (const float*)d_in[0];
  const float* Wih0 = (const float*)d_in[1];
  const float* Whh0 = (const float*)d_in[2];
  const float* bih0 = (const float*)d_in[3];
  const float* bhh0 = (const float*)d_in[4];
  const float* Wih1 = (const float*)d_in[5];
  const float* Whh1 = (const float*)d_in[6];
  const float* bih1 = (const float*)d_in[7];
  const float* bhh1 = (const float*)d_in[8];
  const float* W1   = (const float*)d_in[9];
  const float* b1   = (const float*)d_in[10];
  const float* W2   = (const float*)d_in[11];
  const float* b2   = (const float*)d_in[12];
  float* out = (float*)d_out;

  (void)hipFuncSetAttribute((const void*)gru2_mlp_fused,
                            hipFuncAttributeMaxDynamicSharedMemorySize, SMEM_BYTES);
  gru2_mlp_fused<<<dim3(NBLOCKS), dim3(WAVES * 32), SMEM_BYTES, stream>>>(
      x, Wih0, Whh0, bih0, bhh0, Wih1, Whh1, bih1, bhh1, W1, b1, W2, b2, out);
}
